// Model_14070312862201
// MI455X (gfx1250) — compile-verified, matrix operand order checked by reading, not run on hardware
//
#include <hip/hip_runtime.h>

// GNN (RGCN-style) forward for MI455X / gfx1250.
// Strategy: transform-then-gather (th = h@Wconv, then acc[dst]+=th[src]),
// dense GEMMs via v_wmma_f32_16x16x32_f16 (wave32, 16-row tiles, K=64 in two
// k-steps, N=64 in four 16-col tiles). fp32 accumulation, f16 operands.
// All intermediates (~51 MB) stay L2-resident (192 MB L2).
// f16-out GEMM stages its output tile through wave-private LDS so global
// stores are 4x coalesced global_store_b128 instead of 32x strided b16.

typedef __attribute__((ext_vector_type(16))) _Float16 v16h;
typedef __attribute__((ext_vector_type(8)))  _Float16 v8h;   // 16-byte chunk
typedef __attribute__((ext_vector_type(4)))  _Float16 v4h;   // 8-byte chunk
typedef __attribute__((ext_vector_type(8)))  float    v8f;

#define N_NODES  100000
#define N_LABELS 8
#define N_EDGES  200000
#define IN_FEAT  16
#define NHID     64
#define N_LAYERS 3
#define N_GRAPHS 32
#define WPB      8      // waves per block (256 threads)

union ABFrag { v16h v; v8h c[2]; _Float16 h[16]; };

// A-matrix (16x32 f16, M x K): lane row = lane&15, kbase = (lane>>4)*8.
// elements 0..7 -> K = kbase..kbase+7 ; elements 8..15 -> K = kbase+16..kbase+23
__device__ inline v16h load_a_frag(const _Float16* __restrict__ row, int lane, int ks) {
  ABFrag f;
  const int kbase = ks * 32 + ((lane >> 4) << 3);
  f.c[0] = *(const v8h*)(row + kbase);        // global_load_b128
  f.c[1] = *(const v8h*)(row + kbase + 16);   // global_load_b128
  return f.v;
}

// B-matrix (32x16 f16, K x N), W row-major [64][64]: lane col = nt*16 + (lane&15),
// same K pattern as A. Strided scalar loads; W is 8 KB and cache resident, and
// the fragments live in registers across the whole tile loop.
__device__ inline v16h load_b_frag(const _Float16* __restrict__ W, int lane, int nt, int ks) {
  ABFrag f;
  const int n = nt * 16 + (lane & 15);
  const int kbase = ks * 32 + ((lane >> 4) << 3);
#pragma unroll
  for (int i = 0; i < 8; ++i) f.h[i]     = W[(kbase + i) * NHID + n];
#pragma unroll
  for (int i = 0; i < 8; ++i) f.h[8 + i] = W[(kbase + 16 + i) * NHID + n];
  return f.v;
}

__device__ inline v8f wmma16(v16h a, v16h b, v8f c) {
  return __builtin_amdgcn_wmma_f32_16x16x32_f16(false, a, false, b, (short)0, c, false, false);
}

// C16[rows x 64] = A16[rows x 64] @ W16[64 x 64], f16 output (conv transforms).
// Output staged through wave-private LDS -> 4 coalesced b128 stores per lane.
__global__ void gemm64_f16out(const _Float16* __restrict__ A, const _Float16* __restrict__ W,
                              _Float16* __restrict__ C, int ntiles) {
  __shared__ __align__(16) _Float16 stage[WPB][16 * NHID];   // 2 KB per wave
  const int gtid   = blockIdx.x * blockDim.x + threadIdx.x;
  const int wave   = gtid >> 5;
  const int wloc   = threadIdx.x >> 5;
  const int lane   = threadIdx.x & 31;
  const int nwaves = (gridDim.x * blockDim.x) >> 5;
  _Float16* st = stage[wloc];

  v16h b[4][2];
#pragma unroll
  for (int nt = 0; nt < 4; ++nt)
#pragma unroll
    for (int ks = 0; ks < 2; ++ks)
      b[nt][ks] = load_b_frag(W, lane, nt, ks);

  const int mloc = (lane >> 4) << 3;   // 0 or 8
  const int ncol = lane & 15;
  for (int tile = wave; tile < ntiles; tile += nwaves) {   // wave-uniform: EXEC stays all-1s
    const int row0 = tile << 4;
    const _Float16* arow = A + (size_t)(row0 + (lane & 15)) * NHID;
    const v16h a0 = load_a_frag(arow, lane, 0);
    const v16h a1 = load_a_frag(arow, lane, 1);
#pragma unroll
    for (int nt = 0; nt < 4; ++nt) {
      v8f c = {};
      c = wmma16(a0, b[nt][0], c);
      c = wmma16(a1, b[nt][1], c);
#pragma unroll
      for (int r = 0; r < 8; ++r)
        st[(mloc + r) * NHID + nt * 16 + ncol] = (_Float16)c[r];   // ds_store_b16
    }
    // Wave-private readback (per-wave LDS ops are in-order; no barrier needed),
    // then 4 x global_store_b128 covering 512B-contiguous runs.
    _Float16* crow = C + (size_t)row0 * NHID;
#pragma unroll
    for (int i = 0; i < 4; ++i) {
      const int off = i * 256 + lane * 8;                 // in halves
      *(v8h*)(crow + off) = *(const v8h*)(st + off);      // ds_load_b128 -> global_store_b128
    }
  }
}

// Cf32[rows x 64] = A16 @ W16 + bias  (root transform, f32 out, seeds the accumulator)
__global__ void gemm64_f32bias(const _Float16* __restrict__ A, const _Float16* __restrict__ W,
                               const float* __restrict__ bias, float* __restrict__ C, int ntiles) {
  const int gtid   = blockIdx.x * blockDim.x + threadIdx.x;
  const int wave   = gtid >> 5;
  const int lane   = threadIdx.x & 31;
  const int nwaves = (gridDim.x * blockDim.x) >> 5;

  v16h b[4][2];
  float bv[4];
#pragma unroll
  for (int nt = 0; nt < 4; ++nt) {
    bv[nt] = bias[nt * 16 + (lane & 15)];
#pragma unroll
    for (int ks = 0; ks < 2; ++ks)
      b[nt][ks] = load_b_frag(W, lane, nt, ks);
  }

  for (int tile = wave; tile < ntiles; tile += nwaves) {
    const int row0 = tile << 4;
    const _Float16* arow = A + (size_t)(row0 + (lane & 15)) * NHID;
    const v16h a0 = load_a_frag(arow, lane, 0);
    const v16h a1 = load_a_frag(arow, lane, 1);
    const int mbase = row0 + ((lane >> 4) << 3);
    const int ncol  = lane & 15;
#pragma unroll
    for (int nt = 0; nt < 4; ++nt) {
      v8f c = {};
      c = wmma16(a0, b[nt][0], c);
      c = wmma16(a1, b[nt][1], c);
#pragma unroll
      for (int r = 0; r < 8; ++r)
        C[(size_t)(mbase + r) * NHID + nt * 16 + ncol] = c[r] + bv[nt];
    }
  }
}

// acc[dst[e]][:] += th[src[e]][:]   (16 threads/edge, one b64 gather + 4 f32 atomics;
// th/acc are L2-resident so the random traffic stays on-chip)
__global__ void scatter_edges(const _Float16* __restrict__ th, const int* __restrict__ srcIdx,
                              const int* __restrict__ dstIdx, float* __restrict__ acc) {
  const int t = blockIdx.x * blockDim.x + threadIdx.x;
  const int e = t >> 4;
  if (e >= N_EDGES) return;
  const int part = (t & 15) << 2;
  const int s = srcIdx[e];
  const int d = dstIdx[e];
  const v4h tv = *(const v4h*)(th + (size_t)s * NHID + part);   // global_load_b64
  float* ap = acc + (size_t)d * NHID + part;
#pragma unroll
  for (int i = 0; i < 4; ++i) atomicAdd(ap + i, (float)tv[i]);
}

// h16 = x @ emb_W + emb_b   (K=16, memory-bound; scalar is fine)
__global__ void embed_kernel(const float* __restrict__ x, const float* __restrict__ W,
                             const float* __restrict__ b, _Float16* __restrict__ h) {
  const int t = blockIdx.x * blockDim.x + threadIdx.x;
  if (t >= N_NODES * NHID) return;
  const int n = t >> 6, f = t & 63;
  float s = b[f];
  const float* xr = x + (size_t)n * IN_FEAT;
#pragma unroll
  for (int k = 0; k < IN_FEAT; ++k) s += xr[k] * W[k * NHID + f];
  h[t] = (_Float16)s;
}

__global__ void relu_to_h16(const float* __restrict__ acc, _Float16* __restrict__ h) {
  const int t = blockIdx.x * blockDim.x + threadIdx.x;
  if (t >= N_NODES * NHID) return;
  const float v = acc[t];
  h[t] = (_Float16)(v > 0.f ? v : 0.f);
}

__global__ void f32_to_f16(const float* __restrict__ src, _Float16* __restrict__ dst, int n) {
  const int t = blockIdx.x * blockDim.x + threadIdx.x;
  if (t < n) dst[t] = (_Float16)src[t];
}

__global__ void zero_f32(float* __restrict__ p, int n) {
  const int t = blockIdx.x * blockDim.x + threadIdx.x;
  if (t < n) p[t] = 0.f;
}

__global__ void pool_kernel(const _Float16* __restrict__ h, const int* __restrict__ batch,
                            float* __restrict__ sums, float* __restrict__ counts) {
  const int t = blockIdx.x * blockDim.x + threadIdx.x;
  if (t >= N_NODES * NHID) return;
  const int n = t >> 6, f = t & 63;
  const int g = batch[n];
  atomicAdd(&sums[g * NHID + f], (float)h[t]);
  if (f == 0) atomicAdd(&counts[g], 1.0f);
}

// out[g] = relu(mean_g @ W1 + b1) @ W2 + b2   (32 graphs; one block per graph)
__global__ void final_mlp(const float* __restrict__ sums, const float* __restrict__ counts,
                          const float* __restrict__ W1, const float* __restrict__ b1,
                          const float* __restrict__ W2, const float* __restrict__ b2,
                          float* __restrict__ out) {
  __shared__ float gm[NHID];
  __shared__ float red[NHID];
  const int g = blockIdx.x, f = threadIdx.x;
  float cnt = counts[g];
  cnt = cnt > 1.f ? cnt : 1.f;
  gm[f] = sums[g * NHID + f] / cnt;
  __syncthreads();
  float s = b1[f];
#pragma unroll
  for (int k = 0; k < NHID; ++k) s += gm[k] * W1[k * NHID + f];
  s = s > 0.f ? s : 0.f;
  red[f] = s * W2[f];
  __syncthreads();
  for (int stride = 32; stride > 0; stride >>= 1) {
    if (f < stride) red[f] += red[f + stride];
    __syncthreads();
  }
  if (f == 0) out[g] = red[0] + b2[0];
}

extern "C" void kernel_launch(void* const* d_in, const int* in_sizes, int n_in,
                              void* d_out, int out_size, void* d_ws, size_t ws_size,
                              hipStream_t stream) {
  (void)in_sizes; (void)n_in; (void)out_size; (void)ws_size;
  const float* x          = (const float*)d_in[0];
  const int*   edge_index = (const int*)  d_in[1];   // [8][2][200000]
  const int*   batch      = (const int*)  d_in[2];
  const float* emb_W      = (const float*)d_in[3];
  const float* emb_b      = (const float*)d_in[4];
  const float* Wroot      = (const float*)d_in[5];   // [3][64][64]
  const float* broot      = (const float*)d_in[6];   // [3][64]
  const float* Wconv      = (const float*)d_in[7];   // [3][8][64][64]
  const float* mlp_W1     = (const float*)d_in[8];
  const float* mlp_b1     = (const float*)d_in[9];
  const float* mlp_W2     = (const float*)d_in[10];
  const float* mlp_b2     = (const float*)d_in[11];
  float* out = (float*)d_out;

  // workspace carve-up (256B aligned slices)
  char* ws = (char*)d_ws;
  size_t off = 0;
  auto carve = [&](size_t bytes) { void* p = ws + off; off = (off + bytes + 255) & ~(size_t)255; return p; };
  _Float16* h16     = (_Float16*)carve((size_t)N_NODES * NHID * 2);                    // 12.8 MB
  float*    acc     = (float*)   carve((size_t)N_NODES * NHID * 4);                    // 25.6 MB
  _Float16* th16    = (_Float16*)carve((size_t)N_NODES * NHID * 2);                    // 12.8 MB
  _Float16* w16root = (_Float16*)carve((size_t)N_LAYERS * NHID * NHID * 2);
  _Float16* w16conv = (_Float16*)carve((size_t)N_LAYERS * N_LABELS * NHID * NHID * 2);
  float*    pool    = (float*)   carve((size_t)(N_GRAPHS * NHID + N_GRAPHS) * 4);
  float*    sums    = pool;
  float*    counts  = pool + N_GRAPHS * NHID;

  const int B = 256;
  const int nRoot = N_LAYERS * NHID * NHID;
  const int nConv = N_LAYERS * N_LABELS * NHID * NHID;
  f32_to_f16<<<(nRoot + B - 1) / B, B, 0, stream>>>(Wroot, w16root, nRoot);
  f32_to_f16<<<(nConv + B - 1) / B, B, 0, stream>>>(Wconv, w16conv, nConv);

  embed_kernel<<<(N_NODES * NHID + B - 1) / B, B, 0, stream>>>(x, emb_W, emb_b, h16);

  const int ntiles = N_NODES / 16;                       // 6250, exact
  const int tilesPerWave = 2;                            // amortize B-frag loads
  const int gemmBlocks = (ntiles + WPB * tilesPerWave - 1) / (WPB * tilesPerWave);

  for (int l = 0; l < N_LAYERS; ++l) {
    // acc = h @ Wroot[l] + broot[l]   (seeds accumulator, no memset needed)
    gemm64_f32bias<<<gemmBlocks, B, 0, stream>>>(
        h16, w16root + (size_t)l * NHID * NHID, broot + l * NHID, acc, ntiles);
    for (int lab = 0; lab < N_LABELS; ++lab) {
      // th = h @ Wconv[l][lab]  (transform all nodes once, then gather/scatter)
      gemm64_f16out<<<gemmBlocks, B, 0, stream>>>(
          h16, w16conv + ((size_t)l * N_LABELS + lab) * NHID * NHID, th16, ntiles);
      const int* src = edge_index + (size_t)lab * 2 * N_EDGES;
      const int* dst = src + N_EDGES;
      scatter_edges<<<(N_EDGES * 16 + B - 1) / B, B, 0, stream>>>(th16, src, dst, acc);
    }
    relu_to_h16<<<(N_NODES * NHID + B - 1) / B, B, 0, stream>>>(acc, h16);
  }

  zero_f32<<<(N_GRAPHS * NHID + N_GRAPHS + B - 1) / B, B, 0, stream>>>(pool, N_GRAPHS * NHID + N_GRAPHS);
  pool_kernel<<<(N_NODES * NHID + B - 1) / B, B, 0, stream>>>(h16, batch, sums, counts);
  final_mlp<<<N_GRAPHS, NHID, 0, stream>>>(sums, counts, mlp_W1, mlp_b1, mlp_W2, mlp_b2, out);
}
